// Per_Region_Encoding_35639638622447
// MI455X (gfx1250) — matrix operand order repaired and checked by reading, not compiled, not run on hardware
//
#include <hip/hip_runtime.h>

// ============================================================================
// SPADE-style decoder for MI455X (gfx1250, wave32, WMMA).
//
// All convolutions run as implicit GEMMs on v_wmma_f32_16x16x32_bf16 (bf16
// A/B, f32 accumulate): activations are instance-normalized and weights are
// small-scale gaussians, so bf16 in / f32 accumulate is the right
// precision-throughput point on CDNA5.
//
// ConvTranspose(k3,s2,p1,op1) is decomposed into 4 output parity phases; each
// block handles one phase so only its valid taps enter the K loop.
//
// LDS staging is "fragment-ready": tiles are stored permuted so each lane's
// WMMA fragment is a contiguous 32B run (2x ds_load_b128), per the ISA 16-bit
// A 16x32 / B 32x16 VGPR layouts. Staging uses GLOBAL_LOAD_ASYNC_TO_LDS_B128
// (per-lane LDS dest addresses make the permutation free), guarded by
// __has_builtin with a load+ds_store fallback.
//
// K-loop stages 64 K per barrier pair (8 WMMAs per wave between barriers).
// Workspace requirement: ~128 MB. Determinism: no atomics anywhere.
// ============================================================================

typedef __bf16 bf16;
typedef __attribute__((ext_vector_type(16))) __bf16 v16bf;
typedef __attribute__((ext_vector_type(8)))  float  v8f;
typedef int v4i __attribute__((vector_size(16)));   // builtin's b128 unit type

#if defined(__gfx1250__) && \
    __has_builtin(__builtin_amdgcn_global_load_async_to_lds_b128) && \
    __has_builtin(__builtin_amdgcn_s_wait_asynccnt)
#define USE_ASYNC_LDS 1
#else
#define USE_ASYNC_LDS 0
#endif

// Copy one 16B run global -> LDS (async when available); ok==false writes zeros.
static __device__ __forceinline__ void stage_run16(const bf16* gsrc, bf16* ldst, bool ok)
{
#if USE_ASYNC_LDS
  if (ok) {
    __builtin_amdgcn_global_load_async_to_lds_b128(
        (__attribute__((address_space(1))) v4i*)gsrc,
        (__attribute__((address_space(3))) v4i*)ldst, 0, 0);
  } else {
    *(uint4*)ldst = make_uint4(0u, 0u, 0u, 0u);
  }
#else
  uint4 v = make_uint4(0u, 0u, 0u, 0u);
  if (ok) v = *(const uint4*)gsrc;
  *(uint4*)ldst = v;
#endif
}

static __device__ __forceinline__ void stage_fence()
{
#if USE_ASYNC_LDS
  __builtin_amdgcn_s_wait_asynccnt(0);
#endif
  __syncthreads();
}

// ---------------------------------------------------------------------------
// NCHW f32 -> NHWC bf16 (input conversion)
// ---------------------------------------------------------------------------
__global__ __launch_bounds__(256) void nchw_to_nhwc_kernel(
    const float* __restrict__ src, bf16* __restrict__ dst, int lHW)
{
  long idx = (long)blockIdx.x * 256 + threadIdx.x;   // over N*C*HW (NCHW order)
  int  p   = (int)(idx & ((1 << lHW) - 1));
  int  c   = (int)((idx >> lHW) & 255);
  int  n   = (int)(idx >> (lHW + 8));
  dst[(((long)n << lHW) + p) * 256 + c] = (bf16)src[idx];
}

// ---------------------------------------------------------------------------
// Weight prepack: -> bf16 [tap][co][ci] (co-major so B fragments are
// contiguous-in-ci per lane).
//   swapIO=0: src is (O, I, kh, kw)   (conv3x3 forward, 1x1)
//   swapIO=1: src is (I, O, kh, kw)   (convT weights)
// ---------------------------------------------------------------------------
__global__ __launch_bounds__(256) void prepack_kernel(
    const float* __restrict__ src, bf16* __restrict__ dst, int taps, int swapIO)
{
  long idx = (long)blockIdx.x * 256 + threadIdx.x;   // (tap, co, ci) row-major
  int tap = (int)(idx >> 16);
  int co  = (int)((idx >> 8) & 255);
  int ci  = (int)(idx & 255);
  long s = swapIO ? (((long)ci * 256 + co) * taps + tap)
                  : (((long)co * 256 + ci) * taps + tap);
  dst[idx] = (bf16)src[s];
}

// ---------------------------------------------------------------------------
// Instance norm, deterministic two-stage reduction over HW per (n,c).
// ---------------------------------------------------------------------------
__global__ __launch_bounds__(256) void in_stats1_kernel(
    const bf16* __restrict__ in, float* __restrict__ part, int lHW)
{
  int n = blockIdx.y, c = threadIdx.x;
  int p0 = blockIdx.x << 8;                         // 256 pixels per block
  const bf16* base = in + (((long)n << lHW) << 8) + c;
  float s = 0.f, s2 = 0.f;
  for (int p = p0; p < p0 + 256; ++p) {
    float v = (float)base[(long)p << 8];
    s += v; s2 += v * v;
  }
  long o = (((long)blockIdx.x * gridDim.y + n) * 256 + c) * 2;
  part[o + 0] = s; part[o + 1] = s2;
}

__global__ __launch_bounds__(256) void in_stats2_kernel(
    const float* __restrict__ part, float* __restrict__ sums, int splits)
{
  int n = blockIdx.x, c = threadIdx.x, N = gridDim.x;
  float s = 0.f, s2 = 0.f;
  for (int k = 0; k < splits; ++k) {
    long o = (((long)k * N + n) * 256 + c) * 2;
    s += part[o + 0]; s2 += part[o + 1];
  }
  sums[(n * 256 + c) * 2 + 0] = s;
  sums[(n * 256 + c) * 2 + 1] = s2;
}

__global__ __launch_bounds__(256) void in_apply_kernel(
    const bf16* __restrict__ in, bf16* __restrict__ out,
    const float* __restrict__ sums, int lHW, float invHW)
{
  long idx = (long)blockIdx.x * 256 + threadIdx.x;  // over N*HW*256 (NHWC)
  int c = (int)(idx & 255);
  int n = (int)(idx >> (8 + lHW));
  float m = sums[(n * 256 + c) * 2 + 0] * invHW;
  float v = sums[(n * 256 + c) * 2 + 1] * invHW - m * m;
  out[idx] = (bf16)(((float)in[idx] - m) * rsqrtf(v + 1e-5f));
}

// ---------------------------------------------------------------------------
// Implicit-GEMM conv on WMMA bf16.
//   MODE 0: conv3x3 pad1            EPI 0: +bias, leaky_relu(0.1), store bf16 NHWC
//   MODE 1: convT3x3 s2 (per-phase) EPI 1: +bias, store bf16 NHWC (shortcut)
//                                   EPI 2: +bias, += existing, store (main)
//   MODE 2: conv1x1                 EPI 3: +bias, tanh, store f32 NCHW
//
// Block: 128 output pixels (M) x 64 output channels (N); 8 waves in 4(M)x2(N)
// grid; each wave holds 2x2 16x16 f32 accumulators. K loop: taps x 4 chunks
// of 64 input channels (two WMMA k-steps per barrier pair).
//
// Fragment-ready LDS layouts (per CDNA5 ISA 7.12.2):
//   A (16x32 bf16): lane = (m&15) + 16*((k>>3)&1), elem e = (k&7) + 8*(k>>4)
//     -> per lane, e=0..7 and e=8..15 are contiguous 8-channel runs.
//   B (32x16 bf16): lane = (n&15) + 16*((k>>3)&1), elem e = (k&7) + 8*(k>>4)
//     -> per lane, e=0..7 and e=8..15 are contiguous 8-ci runs at fixed co (weights prepacked co-major).
// ---------------------------------------------------------------------------
template <int MODE, int EPI>
__global__ __launch_bounds__(256) void conv_wmma_kernel(
    const bf16* __restrict__ in,   // NHWC bf16 (Nimg, S, S, 256)
    const bf16* __restrict__ wp,   // bf16 [tap][co][ci]
    const float* __restrict__ bias,
    void* __restrict__ outp,
    int lside)
{
  const int C = 256;
  const int S = 1 << lside;
  const int HW = 1 << (2 * lside);
  const int tid  = threadIdx.x;
  const int lane = tid & 31;
  const int wave = tid >> 5;
  const int wm = wave & 3;          // wave M index (0..3), 32 rows each
  const int wn = wave >> 2;         // wave N index (0..1), 32 cols each

  const int lperImg = 2 * lside + ((MODE == 1) ? 2 : 0);
  const int m0   = blockIdx.x << 7;
  const int nimg = m0 >> lperImg;
  const int mrem = m0 & ((1 << lperImg) - 1);
  const int phase = (MODE == 1) ? (mrem >> (2 * lside)) : 0;
  const int pix0  = mrem & (HW - 1);
  const int pr = phase >> 1, pc = phase & 1;
  const int n0 = blockIdx.y << 6;

  __shared__ __align__(32) bf16 Afrag[2][8][32][16];   // 16 KB (2 k-steps)
  __shared__ __align__(32) bf16 Bfrag[2][4][32][16];   //  8 KB

  const v8f vzero = {0.f, 0.f, 0.f, 0.f, 0.f, 0.f, 0.f, 0.f};
  v8f acc[2][2];
#pragma unroll
  for (int i = 0; i < 2; ++i)
#pragma unroll
    for (int j = 0; j < 2; ++j) acc[i][j] = vzero;

  // A staging: threads 0..127 stage k-step 0, 128..255 stage k-step 1; each
  // thread copies 4 contiguous 16B runs (64B) of one pixel's channels.
  const int a_pl = tid & 127;       // local pixel
  const int a_ks = tid >> 7;        // k-step
  // B staging: 512 runs, 2 per thread.
  const int b_ks = tid >> 7;
  const int b_nn = (tid & 127) >> 1;

  const int KD = (MODE == 2) ? 1 : 3;
  for (int kh = 0; kh < KD; ++kh) {
    if (MODE == 1 && (((kh + pr) & 1) == 0)) continue;   // parity-invalid tap row
    for (int kw = 0; kw < KD; ++kw) {
      if (MODE == 1 && (((kw + pc) & 1) == 0)) continue; // parity-invalid tap col
      const int tap = kh * KD + kw;
      int dh = 0, dw = 0;
      if (MODE == 0) { dh = kh - 1;            dw = kw - 1; }
      if (MODE == 1) { dh = (pr + 1 - kh) >> 1; dw = (pc + 1 - kw) >> 1; } // in {0,1}
      const bf16* wtap = wp + (long)tap * C * C;

      // Per-tap A source row for this thread's pixel (k-chunk invariant).
      int pix = pix0 + a_pl;
      int hh = pix >> lside, ww = pix & (S - 1);
      int ih = hh + dh, iw = ww + dw;
      bool ok = (unsigned)ih < (unsigned)S && (unsigned)iw < (unsigned)S;
      const bf16* arow = in + (((long)nimg * S + ih) * S + iw) * C + a_ks * 32;

      for (int kc = 0; kc < C; kc += 64) {
        // ---- stage A: 1024 16B runs (4 per thread, contiguous) ----
#pragma unroll
        for (int r = 0; r < 4; ++r) {
          stage_run16(arow + kc + r * 8,
                      &Afrag[a_ks][a_pl >> 4][(a_pl & 15) + ((r & 1) << 4)][(r >> 1) << 3],
                      ok);
        }
        // ---- stage B: 512 16B runs (2 per thread) ----
#pragma unroll
        for (int r = 0; r < 2; ++r) {
          int kr = ((tid & 1) << 1) + r;    // run 0..3 within 32-ch step
          stage_run16(wtap + ((long)(n0 + b_nn)) * C + kc + b_ks * 32 + kr * 8,
                      &Bfrag[b_ks][b_nn >> 4][(b_nn & 15) + ((kr & 1) << 4)][(kr >> 1) << 3],
                      true);
        }
        if (kc + 64 < C)
          __builtin_prefetch(wtap + ((long)(n0 + b_nn)) * C + kc + 64 + b_ks * 32, 0, 1);
        stage_fence();   // s_wait_asynccnt 0 (if async) + barrier

        // ---- 8 WMMAs per wave between barriers ----
#pragma unroll
        for (int ks = 0; ks < 2; ++ks) {
          v16bf af[2], bfm[2];
#pragma unroll
          for (int i = 0; i < 2; ++i) af[i]  = *(const v16bf*)&Afrag[ks][wm * 2 + i][lane][0];
#pragma unroll
          for (int j = 0; j < 2; ++j) bfm[j] = *(const v16bf*)&Bfrag[ks][wn * 2 + j][lane][0];
#pragma unroll
          for (int i = 0; i < 2; ++i)
#pragma unroll
            for (int j = 0; j < 2; ++j)
              acc[i][j] = __builtin_amdgcn_wmma_f32_16x16x32_bf16(
                  false, af[i], false, bfm[j], (short)0, acc[i][j], false, false);
        }
        __syncthreads();
      }
    }
  }

  // ---- epilogue: C layout -> lane holds channel co=(lane&15), 8 consecutive
  //      pixels per tile (M = r + 8*(lane>>4)) ----
#pragma unroll
  for (int j = 0; j < 2; ++j) {
    int co = n0 + wn * 32 + j * 16 + (lane & 15);
    float bv = bias[co];
#pragma unroll
    for (int i = 0; i < 2; ++i) {
#pragma unroll
      for (int r = 0; r < 8; ++r) {
        int ml = wm * 32 + i * 16 + r + ((lane >> 4) << 3);
        int pix = pix0 + ml;
        int hh = pix >> lside, ww = pix & (S - 1);
        float v = acc[i][j][r] + bv;
        if (MODE == 0) {
          v = (v > 0.f) ? v : 0.1f * v;   // leaky_relu(0.1)
          ((bf16*)outp)[(((long)nimg * S + hh) * S + ww) * C + co] = (bf16)v;
        } else if (MODE == 1) {
          int oh = 2 * hh + pr, ow = 2 * ww + pc;
          long a = (((long)nimg * (2 * S) + oh) * (2 * S) + ow) * C + co;
          bf16* op = (bf16*)outp;
          if (EPI == 2) v += (float)op[a];   // accumulate onto shortcut
          op[a] = (bf16)v;
        } else {
          ((float*)outp)[((long)(nimg * C + co)) * HW + pix] = tanhf(v);
        }
      }
    }
  }
}

// ---------------------------------------------------------------------------
// Segment pooling: block = (16-channel chunk, segment s in 0..8, batch).
// s==8 computes the global mean row. Deterministic block-local reduction.
// ---------------------------------------------------------------------------
__global__ __launch_bounds__(256) void pool_kernel(
    const float* __restrict__ hbuf,   // (2,256,256,256) NCHW f32
    const float* __restrict__ seg,    // (2,8,256,256) f32
    float* __restrict__ codes,        // (2,9,256)
    float* __restrict__ exist)        // (2,8)
{
  const int HW = 65536;
  int cc = blockIdx.x, s = blockIdx.y, b = blockIdx.z;
  int tid = threadIdx.x;

  float acc[16];
#pragma unroll
  for (int c = 0; c < 16; ++c) acc[c] = 0.f;
  float cnt = 0.f;

  const float* hb = hbuf + ((long)b * 256 + cc * 16) * HW;
  const float* sb = seg + ((long)(b * 8 + s)) * HW;
  for (int p = tid; p < HW; p += 256) {
    float m = 1.f;
    if (s < 8) m = (sb[p] > 0.f) ? 1.f : 0.f;
    cnt += m;
    if (m != 0.f) {
#pragma unroll
      for (int c = 0; c < 16; ++c) acc[c] += hb[(long)c * HW + p];
    }
  }

  __shared__ float red[256];
  red[tid] = cnt; __syncthreads();
  for (int o = 128; o; o >>= 1) { if (tid < o) red[tid] += red[tid + o]; __syncthreads(); }
  float area = red[0];
  __syncthreads();

  for (int c = 0; c < 16; ++c) {
    red[tid] = acc[c]; __syncthreads();
    for (int o = 128; o; o >>= 1) { if (tid < o) red[tid] += red[tid + o]; __syncthreads(); }
    if (tid == 0) {
      float v = (area > 0.f) ? (red[0] / fmaxf(area, 1.f)) : 0.f;
      codes[((long)b * 9 + s) * 256 + cc * 16 + c] = v;
    }
    __syncthreads();
  }
  if (tid == 0 && cc == 0 && s < 8) exist[b * 8 + s] = (area > 0.f) ? 1.f : 0.f;
}

// ---------------------------------------------------------------------------
// Host orchestration
// ---------------------------------------------------------------------------
extern "C" void kernel_launch(void* const* d_in, const int* in_sizes, int n_in,
                              void* d_out, int out_size, void* d_ws, size_t ws_size,
                              hipStream_t stream)
{
  (void)in_sizes; (void)n_in; (void)out_size; (void)ws_size;
  // Input order (setup_inputs dict, params flattened leaf-order):
  //   0:x  1:segmentation  2..19: blocks[i]{w1,b1,w2,b2,ws,bs}  20:wf 21:bf
  const float* x   = (const float*)d_in[0];
  const float* seg = (const float*)d_in[1];
  const float *w1[3], *b1[3], *w2[3], *b2[3], *wsc[3], *bsc[3];
  for (int i = 0; i < 3; ++i) {
    int base = 2 + 6 * i;
    w1[i]  = (const float*)d_in[base + 0]; b1[i]  = (const float*)d_in[base + 1];
    w2[i]  = (const float*)d_in[base + 2]; b2[i]  = (const float*)d_in[base + 3];
    wsc[i] = (const float*)d_in[base + 4]; bsc[i] = (const float*)d_in[base + 5];
  }
  const float* wf  = (const float*)d_in[20];
  const float* bfp = (const float*)d_in[21];

  // Workspace carve-out (~128 MB)
  char* wsb = (char*)d_ws;
  size_t off = 0;
  auto alloc = [&](size_t bytes) -> void* {
    void* p = wsb + off;
    off += (bytes + 255) & ~(size_t)255;
    return p;
  };
  bf16 *wp1[3], *wp2[3], *wps[3];
  for (int i = 0; i < 3; ++i) {
    wp1[i] = (bf16*)alloc((size_t)9 * 65536 * sizeof(bf16));
    wp2[i] = (bf16*)alloc((size_t)9 * 65536 * sizeof(bf16));
    wps[i] = (bf16*)alloc((size_t)9 * 65536 * sizeof(bf16));
  }
  bf16* wpf = (bf16*)alloc((size_t)65536 * sizeof(bf16));
  bf16* xbuf[4];
  for (int i = 0; i < 4; ++i) {
    int S = 32 << i;
    xbuf[i] = (bf16*)alloc((size_t)2 * S * S * 256 * sizeof(bf16));
  }
  bf16*  tn   = (bf16*)alloc((size_t)2 * 128 * 128 * 256 * sizeof(bf16));
  bf16*  tc   = (bf16*)alloc((size_t)2 * 128 * 128 * 256 * sizeof(bf16));
  float* part = (float*)alloc((size_t)64 * 2 * 256 * 2 * sizeof(float));
  float* sums = (float*)alloc((size_t)2 * 256 * 2 * sizeof(float));

  // Weight prepack (bf16, [tap][co][ci])
  for (int i = 0; i < 3; ++i) {
    prepack_kernel<<<dim3(9 * 256), 256, 0, stream>>>(w1[i],  wp1[i], 9, 0);
    prepack_kernel<<<dim3(9 * 256), 256, 0, stream>>>(w2[i],  wp2[i], 9, 1);
    prepack_kernel<<<dim3(9 * 256), 256, 0, stream>>>(wsc[i], wps[i], 9, 1);
  }
  prepack_kernel<<<dim3(256), 256, 0, stream>>>(wf, wpf, 1, 0);

  // Input: NCHW f32 -> NHWC bf16 (32x32)
  nchw_to_nhwc_kernel<<<dim3(2 * 256 * 1024 / 256), 256, 0, stream>>>(x, xbuf[0], 10);

  // Three upsampling residual blocks
  for (int i = 0; i < 3; ++i) {
    int lside = 5 + i;
    int HW = 1 << (2 * lside);
    int splits = HW >> 8;
    float invHW = 1.f / (float)HW;

    // IN(x) -> tn
    in_stats1_kernel<<<dim3(splits, 2), 256, 0, stream>>>(xbuf[i], part, 2 * lside);
    in_stats2_kernel<<<dim3(2), 256, 0, stream>>>(part, sums, splits);
    in_apply_kernel<<<dim3(2 * HW), 256, 0, stream>>>(xbuf[i], tn, sums, 2 * lside, invHW);

    // conv3x3 + bias + leaky_relu -> tc
    conv_wmma_kernel<0, 0><<<dim3(2 * HW / 128, 4), 256, 0, stream>>>(
        tn, wp1[i], b1[i], tc, lside);

    // IN(tc) -> tn (reuse)
    in_stats1_kernel<<<dim3(splits, 2), 256, 0, stream>>>(tc, part, 2 * lside);
    in_stats2_kernel<<<dim3(2), 256, 0, stream>>>(part, sums, splits);
    in_apply_kernel<<<dim3(2 * HW), 256, 0, stream>>>(tc, tn, sums, 2 * lside, invHW);

    // convT shortcut (store) then convT main (accumulate) -> xbuf[i+1]
    conv_wmma_kernel<1, 1><<<dim3(2 * 4 * HW / 128, 4), 256, 0, stream>>>(
        xbuf[i], wps[i], bsc[i], xbuf[i + 1], lside);
    conv_wmma_kernel<1, 2><<<dim3(2 * 4 * HW / 128, 4), 256, 0, stream>>>(
        tn, wp2[i], b2[i], xbuf[i + 1], lside);
  }

  // Final 1x1 conv + bias + tanh -> h (f32 NCHW) directly into d_out
  float* codes = (float*)d_out;
  float* exist = codes + 2 * 9 * 256;
  float* hout  = exist + 2 * 8;
  conv_wmma_kernel<2, 3><<<dim3(2 * 65536 / 128, 4), 256, 0, stream>>>(
      xbuf[3], wpf, bfp, hout, 8);

  // Segment pooling (codes_vector, exist_vector)
  pool_kernel<<<dim3(16, 9, 2), 256, 0, stream>>>(hout, seg, codes, exist);
}
